// MultiHeadAttention_61924838474093
// MI455X (gfx1250) — matrix operand order repaired and checked by reading, not compiled, not run on hardware
//
#include <hip/hip_runtime.h>
#include <hip/hip_bf16.h>

// MHA forward for MI455X (gfx1250, wave32, WMMA bf16).
// d_out = [ output fp32 (B*S*D) | attn fp32 (B*H*S*S) ]

typedef __attribute__((ext_vector_type(16))) __bf16         v16bf;
typedef __attribute__((ext_vector_type(8)))  float          v8f;
typedef __attribute__((ext_vector_type(4)))  float          v4f;
typedef __attribute__((ext_vector_type(16))) unsigned short v16u;
typedef __attribute__((ext_vector_type(8)))  unsigned short v8u;

#define BB_  2
#define S_   2048
#define D_   1024
#define H_   16
#define DK_  64

__device__ __forceinline__ unsigned short f32_to_bf16(float f) {
  unsigned int x = __float_as_uint(f);
  x += 0x7FFFu + ((x >> 16) & 1u);          // round-to-nearest-even
  return (unsigned short)(x >> 16);
}

__device__ __forceinline__ v16bf pack_frag(v8u lo, v8u hi) {
  v16u u;
#pragma unroll
  for (int e = 0; e < 8; ++e) { u[e] = lo[e]; u[e + 8] = hi[e]; }
  return __builtin_bit_cast(v16bf, u);
}

// -------------------- fp32 -> bf16 conversion (vectorized, NT reads) ---------
__global__ __launch_bounds__(256) void cvt_f32_to_bf16(
    const float* __restrict__ in, unsigned short* __restrict__ out, int n4) {
  int i = blockIdx.x * blockDim.x + threadIdx.x;
  if (i < n4) {
    v4f f = __builtin_nontemporal_load((const v4f*)in + i);
    ushort4 o;
    o.x = f32_to_bf16(f.x); o.y = f32_to_bf16(f.y);
    o.z = f32_to_bf16(f.z); o.w = f32_to_bf16(f.w);
    ((ushort4*)out)[i] = o;
  }
}

// -------------------- tiled WMMA GEMM --------------------
// C[M,N] = A[M,K] * B (+ bias), batched over blockIdx.z with linear strides.
// Block: 256 threads = 8 waves (4x2); tile 128x64x64; per-wave 32x32 (2x2 WMMA,
// two K=32 sub-steps per staged tile => 8 WMMA per wave per barrier pair).
// TRANS_B:  B is [N,K] row-major (x @ W^T / q @ k^T);  else B is [K,N].
// A_IS_F32: A read as fp32 non-temporal (attn), converted to bf16 into LDS.
// STORE: 0 = fp32 row-major; 1 = bf16 head-split [B,H,S,dk];
//        2 = bf16 context concat [(b*S+m)*D + h*DK + n].
template <bool TRANS_B, bool A_IS_F32, int STORE, bool HAS_BIAS, bool NT_STORE>
__global__ __launch_bounds__(256) void gemm_wmma(
    const void* __restrict__ Av, const unsigned short* __restrict__ Bmat,
    const float* __restrict__ bias, void* __restrict__ Cv,
    int M, int N, int K, int lda, int ldb, int ldc,
    long long strideA, long long strideB, long long strideC) {
  constexpr int BM = 128, BN = 64, BK = 64, PAD = 8;
  constexpr int BR = TRANS_B ? BN : BK;
  constexpr int BC = TRANS_B ? BK : BN;
  // row stride 72 ushorts = 144 B: 16B-aligned rows, conflict-free frag reads
  __shared__ unsigned short As[BM][BK + PAD];
  __shared__ unsigned short Bs[BR][BC + PAD];

  const int tid  = threadIdx.x;
  const int lane = tid & 31;
  const int wave = tid >> 5;
  const int wm = (wave >> 1) * 32;          // 8 waves as 4(m) x 2(n)
  const int wn = (wave & 1) * 32;
  const int blockM = blockIdx.y * BM;
  const int blockN = blockIdx.x * BN;
  const int z = blockIdx.z;

  const unsigned short* Ab = A_IS_F32 ? nullptr
                                      : ((const unsigned short*)Av + (size_t)z * strideA);
  const float* Af = A_IS_F32 ? ((const float*)Av + (size_t)z * strideA) : nullptr;
  const unsigned short* Bb = Bmat + (size_t)z * strideB;

  v8f acc[2][2] = {};

  for (int k0 = 0; k0 < K; k0 += BK) {
    // ---- stage A tile (BM x BK): 8192 elems, 32/thread ----
#pragma unroll
    for (int c = 0; c < 4; ++c) {
      int idx = (c * 256 + tid) * 8;
      int r = idx >> 6, col = idx & 63;
      if constexpr (A_IS_F32) {
        const float* src = Af + (size_t)(blockM + r) * lda + k0 + col;
        v4f f0 = __builtin_nontemporal_load((const v4f*)src);
        v4f f1 = __builtin_nontemporal_load((const v4f*)(src + 4));
        union { unsigned short s[8]; uint4 v; } u;
        u.s[0] = f32_to_bf16(f0.x); u.s[1] = f32_to_bf16(f0.y);
        u.s[2] = f32_to_bf16(f0.z); u.s[3] = f32_to_bf16(f0.w);
        u.s[4] = f32_to_bf16(f1.x); u.s[5] = f32_to_bf16(f1.y);
        u.s[6] = f32_to_bf16(f1.z); u.s[7] = f32_to_bf16(f1.w);
        *(uint4*)&As[r][col] = u.v;
        __builtin_prefetch(src + BK, 0, 1);      // speculative: safe past end
      } else {
        const unsigned short* src = Ab + (size_t)(blockM + r) * lda + k0 + col;
        *(uint4*)&As[r][col] = *(const uint4*)src;
        __builtin_prefetch(src + BK, 0, 1);
      }
    }
    // ---- stage B tile: 4096 elems, 16/thread ----
#pragma unroll
    for (int c = 0; c < 2; ++c) {
      int idx = (c * 256 + tid) * 8;
      int r = idx >> 6, col = idx & 63;
      if constexpr (TRANS_B) {                   // [BN][BK], K-contiguous
        const unsigned short* src = Bb + (size_t)(blockN + r) * ldb + k0 + col;
        *(uint4*)&Bs[r][col] = *(const uint4*)src;
        __builtin_prefetch(src + BK, 0, 1);
      } else {                                   // [BK][BN], N-contiguous
        const unsigned short* src = Bb + (size_t)(k0 + r) * ldb + blockN + col;
        *(uint4*)&Bs[r][col] = *(const uint4*)src;
        __builtin_prefetch(src + (size_t)BK * ldb, 0, 1);
      }
    }
    __syncthreads();

    // ---- two K=32 sub-steps over the staged tile ----
#pragma unroll
    for (int ks = 0; ks < BK; ks += 32) {
      v16bf afrag[2], bfrag[2];
#pragma unroll
      for (int i = 0; i < 2; ++i) {
        // A 16x32 bf16: lanes 0-15 -> K 0..15; lanes 16-31 -> K 16..31
        int r  = wm + i * 16 + (lane & 15);
        int kb = ks + ((lane >> 4) << 4);
        const unsigned short* p = &As[r][kb];
        afrag[i] = pack_frag(*(const v8u*)p, *(const v8u*)(p + 8));
      }
#pragma unroll
      for (int j = 0; j < 2; ++j) {
        // B 32x16 bf16: lane n = lane&15, K = e + 16*(lane>=16)
        int n  = wn + j * 16 + (lane & 15);
        int kb = ks + ((lane >> 4) << 4);
        if constexpr (TRANS_B) {
          const unsigned short* p = &Bs[n][kb];
          bfrag[j] = pack_frag(*(const v8u*)p, *(const v8u*)(p + 8));
        } else {
          v16u u;
#pragma unroll
          for (int e = 0; e < 16; ++e) u[e] = Bs[kb + e][n];
          bfrag[j] = __builtin_bit_cast(v16bf, u);
        }
      }
#pragma unroll
      for (int i = 0; i < 2; ++i)
#pragma unroll
        for (int j = 0; j < 2; ++j)
          acc[i][j] = __builtin_amdgcn_wmma_f32_16x16x32_bf16(
              false, afrag[i], false, bfrag[j], (short)0, acc[i][j], false, false);
    }
    __syncthreads();
  }

  // ---- epilogue: C layout VGPR r -> M = r + 8*(lane>=16), N = lane&15 ----
#pragma unroll
  for (int i = 0; i < 2; ++i) {
#pragma unroll
    for (int j = 0; j < 2; ++j) {
      const int tm = blockM + wm + i * 16 + ((lane >> 4) << 3);
      const int tn = blockN + wn + j * 16 + (lane & 15);
      float badd = 0.0f;
      if constexpr (HAS_BIAS) badd = bias[tn];     // one load per tile per lane
      if constexpr (STORE == 0) {
        float* dst = (float*)Cv + (size_t)z * strideC + (size_t)tm * ldc + tn;
#pragma unroll
        for (int r = 0; r < 8; ++r) {
          float val = acc[i][j][r] + badd;
          if constexpr (NT_STORE) __builtin_nontemporal_store(val, dst);
          else                    *dst = val;
          dst += ldc;
        }
      } else if constexpr (STORE == 1) {
        // tm multiple of 8 -> rows tm..tm+7 share batch; head h = tn>>6
        unsigned short* dst = (unsigned short*)Cv +
            (size_t)(tm >> 11) * ((size_t)H_ * S_ * DK_) +
            (size_t)(tn >> 6) * ((size_t)S_ * DK_) +
            (size_t)(tm & (S_ - 1)) * DK_ + (size_t)(tn & (DK_ - 1));
#pragma unroll
        for (int r = 0; r < 8; ++r) { dst[0] = f32_to_bf16(acc[i][j][r] + badd); dst += DK_; }
      } else {
        const int b = z >> 4, h = z & 15;
        unsigned short* dst = (unsigned short*)Cv +
            ((size_t)(b * S_ + tm)) * D_ + (size_t)h * DK_ + tn;
#pragma unroll
        for (int r = 0; r < 8; ++r) { dst[0] = f32_to_bf16(acc[i][j][r] + badd); dst += D_; }
      }
    }
  }
}

// -------------------- row softmax (scale folded in, NT traffic) --------------
__global__ __launch_bounds__(256) void softmax_rows(float* __restrict__ attn) {
  __shared__ float red[256];
  const float scale = 0.125f;               // 1/sqrt(d_k=64)
  float* p = attn + (size_t)blockIdx.x * S_;
  const int t = threadIdx.x;
  v4f v[2];
  float mx = -3.402823466e38f;
#pragma unroll
  for (int j = 0; j < 2; ++j) {
    v[j] = __builtin_nontemporal_load((const v4f*)(p + j * 1024) + t) * scale;
#pragma unroll
    for (int e = 0; e < 4; ++e) mx = fmaxf(mx, v[j][e]);
  }
  red[t] = mx;
  __syncthreads();
  for (int s = 128; s > 0; s >>= 1) {
    if (t < s) red[t] = fmaxf(red[t], red[t + s]);
    __syncthreads();
  }
  const float rowMax = red[0];
  __syncthreads();
  float sum = 0.f;
#pragma unroll
  for (int j = 0; j < 2; ++j)
#pragma unroll
    for (int e = 0; e < 4; ++e) { v[j][e] = __expf(v[j][e] - rowMax); sum += v[j][e]; }
  red[t] = sum;
  __syncthreads();
  for (int s = 128; s > 0; s >>= 1) {
    if (t < s) red[t] += red[t + s];
    __syncthreads();
  }
  const float inv = 1.0f / red[0];
#pragma unroll
  for (int j = 0; j < 2; ++j)
    __builtin_nontemporal_store(v[j] * inv, (v4f*)(p + j * 1024) + t);
}

// -------------------- host launcher --------------------
extern "C" void kernel_launch(void* const* d_in, const int* in_sizes, int n_in,
                              void* d_out, int out_size, void* d_ws, size_t ws_size,
                              hipStream_t stream) {
  (void)in_sizes; (void)n_in; (void)out_size; (void)ws_size;
  const float* Q  = (const float*)d_in[0];
  const float* Km = (const float*)d_in[1];
  const float* V  = (const float*)d_in[2];
  const float* Wq = (const float*)d_in[3];
  const float* bq = (const float*)d_in[4];
  const float* Wk = (const float*)d_in[5];
  const float* bk = (const float*)d_in[6];
  const float* Wv = (const float*)d_in[7];
  const float* bv = (const float*)d_in[8];
  const float* Wo = (const float*)d_in[9];
  const float* bo = (const float*)d_in[10];

  float* out  = (float*)d_out;
  float* attn = out + (size_t)BB_ * S_ * D_;

  const size_t SZ_X = (size_t)BB_ * S_ * D_;   // 4 Mi elems
  const size_t SZ_W = (size_t)D_ * D_;         // 1 Mi elems
  unsigned short* ws  = (unsigned short*)d_ws;
  unsigned short* Qb  = ws;
  unsigned short* Kb  = Qb  + SZ_X;
  unsigned short* Vb  = Kb  + SZ_X;
  unsigned short* Wqb = Vb  + SZ_X;
  unsigned short* Wkb = Wqb + SZ_W;
  unsigned short* Wvb = Wkb + SZ_W;
  unsigned short* Wob = Wvb + SZ_W;
  unsigned short* qh  = Wob + SZ_W;            // [B,H,S,dk] bf16
  unsigned short* kh  = qh  + SZ_X;
  unsigned short* vh  = kh  + SZ_X;
  unsigned short* ctx = vh  + SZ_X;            // [B*S, D] bf16

  // 1) fp32 -> bf16 conversions
  cvt_f32_to_bf16<<<(int)(SZ_X / 1024), 256, 0, stream>>>(Q,  Qb,  (int)(SZ_X / 4));
  cvt_f32_to_bf16<<<(int)(SZ_X / 1024), 256, 0, stream>>>(Km, Kb,  (int)(SZ_X / 4));
  cvt_f32_to_bf16<<<(int)(SZ_X / 1024), 256, 0, stream>>>(V,  Vb,  (int)(SZ_X / 4));
  cvt_f32_to_bf16<<<(int)(SZ_W / 1024), 256, 0, stream>>>(Wq, Wqb, (int)(SZ_W / 4));
  cvt_f32_to_bf16<<<(int)(SZ_W / 1024), 256, 0, stream>>>(Wk, Wkb, (int)(SZ_W / 4));
  cvt_f32_to_bf16<<<(int)(SZ_W / 1024), 256, 0, stream>>>(Wv, Wvb, (int)(SZ_W / 4));
  cvt_f32_to_bf16<<<(int)(SZ_W / 1024), 256, 0, stream>>>(Wo, Wob, (int)(SZ_W / 4));

  dim3 blk(256);
  // 2) projections: [4096,1024] @ W^T + b -> head-split bf16
  dim3 gProj(D_ / 64, (BB_ * S_) / 128, 1);
  gemm_wmma<true, false, 1, true, false><<<gProj, blk, 0, stream>>>(
      Qb, Wqb, bq, qh, BB_ * S_, D_, D_, D_, D_, 0, 0, 0, 0);
  gemm_wmma<true, false, 1, true, false><<<gProj, blk, 0, stream>>>(
      Kb, Wkb, bk, kh, BB_ * S_, D_, D_, D_, D_, 0, 0, 0, 0);
  gemm_wmma<true, false, 1, true, false><<<gProj, blk, 0, stream>>>(
      Vb, Wvb, bv, vh, BB_ * S_, D_, D_, D_, D_, 0, 0, 0, 0);

  // 3) scores = q @ k^T per (b,h): raw fp32 into attn region (NT stores)
  dim3 gSc(S_ / 64, S_ / 128, BB_ * H_);
  gemm_wmma<true, false, 0, false, true><<<gSc, blk, 0, stream>>>(
      qh, kh, nullptr, attn, S_, S_, DK_, DK_, DK_, S_,
      (long long)S_ * DK_, (long long)S_ * DK_, (long long)S_ * S_);

  // 4) softmax rows (scale folded in), in place
  softmax_rows<<<BB_ * H_ * S_, 256, 0, stream>>>(attn);

  // 5) out_h = attn @ v per (b,h): A read fp32 NT, context stored bf16 concat
  dim3 gAV(1, S_ / 128, BB_ * H_);
  gemm_wmma<false, true, 2, false, false><<<gAV, blk, 0, stream>>>(
      attn, vh, nullptr, ctx, S_, DK_, S_, S_, DK_, 0,
      (long long)S_ * S_, (long long)S_ * DK_, 0);

  // 6) output projection -> fp32 out
  dim3 gO(D_ / 64, (BB_ * S_) / 128, 1);
  gemm_wmma<true, false, 0, true, false><<<gO, blk, 0, stream>>>(
      ctx, Wob, bo, out, BB_ * S_, D_, D_, D_, D_, D_, 0, 0, 0);
}
